// ResidualVectorQuantizer_42125039239720
// MI455X (gfx1250) — matrix operand order repaired and checked by reading, not compiled, not run on hardware
//
#include <hip/hip_runtime.h>
#include <hip/hip_bf16.h>

// ---------------- problem constants (from reference) ----------------
#define LEVELS   4
#define KCODES   2048
#define DIM      512
#define NROWS    32768                  // 16 * 2048 residual vectors
#define NUMEL    (NROWS * DIM)          // 16,777,216 elements
#define BETA     0.25f

// ---------------- WMMA tiling ----------------
#define ROWS_PER_BLOCK 128              // 8 waves x 16 rows
#define CODE_TILE      64               // codebook rows staged in LDS per stage
#define PITCH          (DIM + 8)        // bf16 elements per LDS row (bank stagger)
#define CHUNKS_PER_TILE (CODE_TILE * DIM / 8)   // 16-byte chunks per B tile

typedef __attribute__((ext_vector_type(16))) __bf16 v16bf;
typedef __attribute__((ext_vector_type(8)))  __bf16 v8bf;
typedef __attribute__((ext_vector_type(4)))  __bf16 v4bf;
typedef __attribute__((ext_vector_type(8)))  float  v8f;
typedef __attribute__((ext_vector_type(4)))  int    v4i;
typedef __attribute__((address_space(1)))    v4i    gv4i;   // global 16B chunk
typedef __attribute__((address_space(3)))    v4i    lv4i;   // LDS 16B chunk

#if __has_builtin(__builtin_amdgcn_global_load_async_to_lds_b128)
#define ASYNC_STAGE 1
#else
#define ASYNC_STAGE 0
#endif

__device__ __forceinline__ void wait_async0() {
#if __has_builtin(__builtin_amdgcn_s_wait_asynccnt)
    __builtin_amdgcn_s_wait_asynccnt(0);
#else
    asm volatile("s_wait_asynccnt 0" ::: "memory");
#endif
}

// Build a 16-element bf16 fragment from two 16-byte loads (LDS or global).
__device__ __forceinline__ v16bf load_frag(const __bf16* p0, const __bf16* p1) {
    v8bf lo = *(const v8bf*)p0;
    v8bf hi = *(const v8bf*)p1;
    v16bf r;
#pragma unroll
    for (int i = 0; i < 8; ++i) { r[i] = lo[i]; r[i + 8] = hi[i]; }
    return r;
}

// ---------------- kernel 0: f32 -> bf16 bulk convert ----------------
__global__ __launch_bounds__(256)
void conv_f32_bf16(const float* __restrict__ src, __bf16* __restrict__ dst) {
    size_t e = ((size_t)blockIdx.x * 256 + threadIdx.x) * 4;
    float4 v = *(const float4*)(src + e);
    v4bf o;
    o[0] = (__bf16)v.x; o[1] = (__bf16)v.y; o[2] = (__bf16)v.z; o[3] = (__bf16)v.w;
    *(v4bf*)(dst + e) = o;
}

// ---------------- kernel 1: codebook row norms ----------------
__global__ __launch_bounds__(256)
void rvq_norms(const float* __restrict__ codebooks, float* __restrict__ cnorm) {
    int lane = threadIdx.x & 31;
    int wave = threadIdx.x >> 5;
    int row  = blockIdx.x * 8 + wave;              // [0, LEVELS*KCODES)
    const float* w = codebooks + (size_t)row * DIM;
    float s = 0.f;
#pragma unroll
    for (int c = lane; c < DIM; c += 32) { float v = w[c]; s += v * v; }
#pragma unroll
    for (int m = 16; m > 0; m >>= 1) s += __shfl_xor(s, m, 32);
    if (lane == 0) cnorm[row] = s;
}

__global__ void rvq_init(float* loss_acc) { *loss_acc = 0.f; }

// ---------------- B-tile staging (async double-buffer or sync fallback) ------
__device__ __forceinline__ void stage_tile(const __bf16* __restrict__ gsrc,
                                           __bf16* __restrict__ bs, int tid) {
#pragma unroll
    for (int i = 0; i < CHUNKS_PER_TILE / 256; ++i) {
        int ch = tid + i * 256;
        int r = ch >> 6;               // 64 chunks (of 8 bf16) per 512-elem row
        int c = (ch & 63) * 8;
#if ASYNC_STAGE
        __builtin_amdgcn_global_load_async_to_lds_b128(
            (gv4i*)(gsrc + (size_t)r * DIM + c),
            (lv4i*)(bs + r * PITCH + c),
            0, 0);
#else
        *(v8bf*)(bs + r * PITCH + c) = *(const v8bf*)(gsrc + (size_t)r * DIM + c);
#endif
    }
}

// ---------------- kernel 2: distance GEMM + argmin (WMMA bf16) ----------------
__global__ __launch_bounds__(256)
void rvq_argmin(const __bf16* __restrict__ rbf,       // [NROWS, DIM] bf16 residual
                const __bf16* __restrict__ cbbf,      // [KCODES, DIM] bf16 codebook
                const float*  __restrict__ cnorm,     // [KCODES]
                int* __restrict__ codes)              // [NROWS]
{
    extern __shared__ __bf16 smem[];                  // 2 x CODE_TILE x PITCH

    const int tid     = threadIdx.x;
    const int lane    = tid & 31;
    const int wave    = tid >> 5;
    const int rowbase = blockIdx.x * ROWS_PER_BLOCK;

    // fragment addressing: contiguous k order (k = 16*half + s), same for A and B (wave32)
    const int halfSel = (lane >= 16);
    const int klocalA = halfSel ? 16 : 0;    // A: lanes 16-31 hold K=16..31
    const int khalfB  = halfSel ? 16 : 0;    // B: lanes 16-31 hold K=16..31
    const int colB    = lane & 15;           // B/D column within the 16-wide tile

    // ---- A fragments: this wave's 16 rows, K=512, held in 128 VGPRs ----
    const __bf16* arow = rbf + (size_t)(rowbase + wave * 16 + (lane & 15)) * DIM;
    v16bf afrag[16];
#pragma unroll
    for (int f = 0; f < 16; ++f) {
        int kb = f * 32 + klocalA;
        afrag[f] = load_frag(arow + kb, arow + kb + 8);
    }

    // per-lane running min/argmin for the 8 C/D accumulator slots
    float minv[8]; int mini[8];
#pragma unroll
    for (int g = 0; g < 8; ++g) { minv[g] = 3.4e38f; mini[g] = 0; }

    // prologue: stage first tile into buffer 0
    stage_tile(cbbf, smem, tid);
#if ASYNC_STAGE
    wait_async0();
#endif
    __syncthreads();

    for (int ct = 0; ct < KCODES; ct += CODE_TILE) {
        const int cur = (ct / CODE_TILE) & 1;
        __bf16* bs   = smem + cur * (CODE_TILE * PITCH);
        __bf16* bnxt = smem + (1 - cur) * (CODE_TILE * PITCH);
        if (ct + CODE_TILE < KCODES)      // overlap next-tile DMA with compute
            stage_tile(cbbf + (size_t)(ct + CODE_TILE) * DIM, bnxt, tid);

        // ---- two independent accumulator chains per k-sweep ----
#pragma unroll
        for (int cbp = 0; cbp < CODE_TILE; cbp += 32) {
            v8f acc0 = {}, acc1 = {};
            const __bf16* b0 = bs + (cbp + colB) * PITCH + khalfB;
            const __bf16* b1 = b0 + 16 * PITCH;
#pragma unroll
            for (int kk = 0; kk < DIM; kk += 32) {
                v16bf f0 = load_frag(b0 + kk, b0 + kk + 8);
                v16bf f1 = load_frag(b1 + kk, b1 + kk + 8);
                acc0 = __builtin_amdgcn_wmma_f32_16x16x32_bf16(
                           false, afrag[kk >> 5], false, f0, (short)0, acc0, false, false);
                acc1 = __builtin_amdgcn_wmma_f32_16x16x32_bf16(
                           false, afrag[kk >> 5], false, f1, (short)0, acc1, false, false);
            }
            const int g0 = ct + cbp + colB, g1 = g0 + 16;
            const float cn0 = cnorm[g0], cn1 = cnorm[g1];
#pragma unroll
            for (int g = 0; g < 8; ++g) {
                float d0 = cn0 - 2.0f * acc0[g];     // ||r||^2 omitted: row-constant
                if (d0 < minv[g]) { minv[g] = d0; mini[g] = g0; }
                float d1 = cn1 - 2.0f * acc1[g];
                if (d1 < minv[g]) { minv[g] = d1; mini[g] = g1; }
            }
        }

#if ASYNC_STAGE
        if (ct + CODE_TILE < KCODES) wait_async0();
#endif
        __syncthreads();   // tile consumed by all waves; next tile visible
    }

    // ---- cross-lane argmin over the 16 columns (groups of 16 lanes) ----
#pragma unroll
    for (int g = 0; g < 8; ++g) {
#pragma unroll
        for (int m = 1; m < 16; m <<= 1) {
            float ov = __shfl_xor(minv[g], m, 32);
            int   oi = __shfl_xor(mini[g], m, 32);
            if (ov < minv[g] || (ov == minv[g] && oi < mini[g])) {
                minv[g] = ov; mini[g] = oi;
            }
        }
    }
    if ((lane & 15) == 0) {
        int rbase = rowbase + wave * 16 + (halfSel ? 8 : 0);
#pragma unroll
        for (int g = 0; g < 8; ++g) codes[rbase + g] = mini[g];
    }
}

// ---------------- kernel 3: gather, residual/quantized update, loss ----------------
__global__ __launch_bounds__(256)
void rvq_update(const float* __restrict__ rin, const float* __restrict__ codebook,
                const int* __restrict__ codes, float* __restrict__ quantized,
                float* __restrict__ rout, __bf16* __restrict__ rbf_out,
                float* __restrict__ loss_acc, int level)
{
    __shared__ float wsum[8];
    int e    = blockIdx.x * 256 + threadIdx.x;     // element index
    int lane = threadIdx.x & 31;
    int wave = threadIdx.x >> 5;
    int row  = e >> 9, c = e & (DIM - 1);

    int   idx  = codes[row];
    float q    = codebook[(size_t)idx * DIM + c];
    float r    = rin[e];
    float diff = r - q;
    quantized[e] = (level == 0) ? q : quantized[e] + q;
    rout[e]      = diff;
    rbf_out[e]   = (__bf16)diff;                   // bf16 copy for next level's GEMM

    float v = diff * diff;
#pragma unroll
    for (int m = 16; m > 0; m >>= 1) v += __shfl_xor(v, m, 32);
    if (lane == 0) wsum[wave] = v;
    __syncthreads();
    if (wave == 0) {
        float s = (lane < 8) ? wsum[lane] : 0.f;
#pragma unroll
        for (int m = 4; m > 0; m >>= 1) s += __shfl_xor(s, m, 32);
        if (lane == 0) atomicAdd(loss_acc, s);
    }
}

__global__ void rvq_finalize(const float* loss_acc, float* loss_out) {
    *loss_out = loss_acc[0] * (1.0f + BETA) / (float)NUMEL;
}

// ---------------- host-side launch ----------------
extern "C" void kernel_launch(void* const* d_in, const int* in_sizes, int n_in,
                              void* d_out, int out_size, void* d_ws, size_t ws_size,
                              hipStream_t stream) {
    const float* x         = (const float*)d_in[0];    // [16,2048,512]
    const float* codebooks = (const float*)d_in[1];    // [4,2048,512]

    // d_out tuple layout: quantized (fp32, NUMEL) | codes (i32, 4*NROWS) | loss (1)
    float* quant    = (float*)d_out;
    int*   codes    = (int*)(quant + NUMEL);
    float* loss_out = (float*)d_out + NUMEL + LEVELS * NROWS;

    // workspace: cnorm[8192] | loss_acc | cbbf (bf16, 8MB) | rbf (bf16, 32MB) | resid f32 (64MB)
    char*   ws       = (char*)d_ws;
    float*  cnorm    = (float*)ws;                                   // 32 KB
    float*  loss_acc = cnorm + LEVELS * KCODES;
    __bf16* cbbf     = (__bf16*)(ws + (64 << 10));                   // 8 MB
    __bf16* rbf      = (__bf16*)(ws + (64 << 10) + ((size_t)8 << 20));   // 32 MB
    float*  resid    = (float*)(ws + (64 << 10) + ((size_t)40 << 20));   // 64 MB

    rvq_norms<<<LEVELS * KCODES / 8, 256, 0, stream>>>(codebooks, cnorm);
    rvq_init<<<1, 1, 0, stream>>>(loss_acc);
    conv_f32_bf16<<<LEVELS * KCODES * DIM / 1024, 256, 0, stream>>>(codebooks, cbbf);
    conv_f32_bf16<<<NUMEL / 1024, 256, 0, stream>>>(x, rbf);         // bf16(x) = level-0 residual

    const size_t lds = (size_t)2 * CODE_TILE * PITCH * sizeof(__bf16);  // ~130 KB
    for (int l = 0; l < LEVELS; ++l) {
        const float* rin = (l == 0) ? x : resid;
        const float* W   = codebooks + (size_t)l * KCODES * DIM;
        rvq_argmin<<<NROWS / ROWS_PER_BLOCK, 256, lds, stream>>>(
            rbf, cbbf + (size_t)l * KCODES * DIM, cnorm + l * KCODES, codes + l * NROWS);
        rvq_update<<<NUMEL / 256, 256, 0, stream>>>(
            rin, W, codes + l * NROWS, quant, resid, rbf, loss_acc, l);
    }
    rvq_finalize<<<1, 1, 0, stream>>>(loss_acc, loss_out);
}
